// CausalSelfAttention_83408264888928
// MI455X (gfx1250) — matrix operand order repaired and checked by reading, not compiled, not run on hardware
//
#include <hip/hip_runtime.h>
#include <hip/hip_bf16.h>

// ---------------- types ----------------
typedef __attribute__((ext_vector_type(16))) __bf16        v16bf;
typedef __attribute__((ext_vector_type(8)))  float         v8f;
typedef __attribute__((ext_vector_type(4)))  unsigned int  u32x4;
typedef __attribute__((ext_vector_type(4)))  int           v4i;

#define B_   2
#define T_   2048
#define C_   1024
#define H_   16
#define D_   64
#define MTOT (B_*T_)            // 4096
#define NQKV (3*C_)             // 3072
#define QKV1 (B_*H_*T_*D_)      // 4194304 elements per q/k/v tensor

// ---------------- CDNA5 async global->LDS path (guarded) ----------------
#if defined(__has_builtin)
#  if __has_builtin(__builtin_amdgcn_global_load_async_to_lds_b128) && \
      __has_builtin(__builtin_amdgcn_s_wait_asynccnt)
#    define USE_ASYNC_LDS 1
#  endif
#endif
#ifndef USE_ASYNC_LDS
#  define USE_ASYNC_LDS 0
#endif

typedef __attribute__((address_space(1))) v4i g_v4i;
typedef __attribute__((address_space(3))) v4i l_v4i;

// per-lane 16B copy global -> LDS
__device__ __forceinline__ void cp16(const unsigned short* g, unsigned short* l) {
#if USE_ASYNC_LDS
    __builtin_amdgcn_global_load_async_to_lds_b128(
        (g_v4i*)g, (l_v4i*)l, 0, 0);
#else
    *(u32x4*)l = *(const u32x4*)g;
#endif
}
__device__ __forceinline__ void async_join() {
#if USE_ASYNC_LDS
    __builtin_amdgcn_s_wait_asynccnt(0);
#endif
}

// ---------------- helpers ----------------
__device__ __forceinline__ unsigned short f2bf(float f) {
    unsigned int u = __builtin_bit_cast(unsigned int, f);
    unsigned int r = u + 0x7FFFu + ((u >> 16) & 1u);   // round-to-nearest-even
    return (unsigned short)(r >> 16);
}

__device__ __forceinline__ v8f vzero() {
    v8f z = {0.f,0.f,0.f,0.f,0.f,0.f,0.f,0.f};
    return z;
}

// load a 16-element bf16 fragment from two 16B-aligned 8-element chunks
__device__ __forceinline__ v16bf ldfrag(const unsigned short* p0, const unsigned short* p1) {
    union { v16bf v; u32x4 q[2]; } f;
    f.q[0] = *(const u32x4*)p0;
    f.q[1] = *(const u32x4*)p1;
    return f.v;
}

__device__ __forceinline__ v8f wmma_bf16(v16bf a, v16bf b, v8f c) {
    return __builtin_amdgcn_wmma_f32_16x16x32_bf16(false, a, false, b, (short)0, c, false, false);
}

__device__ __forceinline__ float bcast(float x, int srcLane) {
    return __builtin_bit_cast(float,
        __builtin_amdgcn_ds_bpermute(srcLane * 4, __builtin_bit_cast(int, x)));
}

// ---------------- fp32 -> bf16 elementwise ----------------
__global__ __launch_bounds__(256)
void cvt_f32_bf16(const float* __restrict__ src, unsigned short* __restrict__ dst, int n) {
    int i = blockIdx.x * blockDim.x + threadIdx.x;
    int stride = gridDim.x * blockDim.x;
    for (; i < n; i += stride) dst[i] = f2bf(src[i]);
}

// ---------------- fp32 [K][N] -> bf16 [N][K] tiled transpose ----------------
__global__ __launch_bounds__(256)
void cvt_transpose(const float* __restrict__ src, unsigned short* __restrict__ dst,
                   int K, int N) {
    __shared__ __align__(16) unsigned short tile[64][72];
    const int k0 = blockIdx.y * 64, n0 = blockIdx.x * 64;
    const int t = threadIdx.x;
    const int r = t >> 2, seg = (t & 3) * 16;
    // coalesced read: 16 consecutive n of row k0+r
    const float* s = src + (size_t)(k0 + r) * N + n0 + seg;
    #pragma unroll
    for (int i = 0; i < 16; ++i) tile[r][seg + i] = f2bf(s[i]);
    __syncthreads();
    // coalesced write: 16 consecutive k of row n0+r
    union { u32x4 q[2]; unsigned short u[16]; } o;
    #pragma unroll
    for (int i = 0; i < 16; ++i) o.u[i] = tile[seg + i][r];
    unsigned short* dp = dst + (size_t)(n0 + r) * K + k0 + seg;
    *(u32x4*)dp = o.q[0];
    *(u32x4*)(dp + 8) = o.q[1];
}

// ---------------- GEMM tile engine ----------------
// C[128x256] tile of A[M x 1024](bf16 row-major) * Wt[N x 1024](bf16, pre-transposed).
// 256 threads = 8 waves in 2(M) x 4(N); each wave 64x64 = acc[4][4] of 16x16.
// LDS (double-buffered): A [128][32] stride 40; B [256][32] stride 40 (both [row][k]).

__device__ __forceinline__ void gemm_stage(const unsigned short* __restrict__ A,
                                           const unsigned short* __restrict__ Wt,
                                           int m0, int n0, int kb,
                                           unsigned short* Al, unsigned short* Bl) {
    const int tid  = threadIdx.x;
    const int arow = tid >> 1;          // 0..127
    const int aseg = (tid & 1) * 16;    // 0/16 ushorts (32B halves)
    const unsigned short* ga = A + (size_t)(m0 + arow) * 1024 + kb + aseg;
    cp16(ga,     &Al[arow * 40 + aseg]);
    cp16(ga + 8, &Al[arow * 40 + aseg + 8]);
    // B: one n-row (32 k = 64B) per thread
    const unsigned short* gb = Wt + (size_t)(n0 + tid) * 1024 + kb;
    unsigned short* lb = &Bl[tid * 40];
    cp16(gb,      lb);
    cp16(gb + 8,  lb + 8);
    cp16(gb + 16, lb + 16);
    cp16(gb + 24, lb + 24);
}

__device__ __forceinline__ void gemm_compute(const unsigned short* Al,
                                             const unsigned short* Bl,
                                             v8f acc[4][4]) {
    const int lane = threadIdx.x & 31;
    const int wave = threadIdx.x >> 5;
    const int wm = wave >> 2, wn = wave & 3;
    const int half = lane >> 4, l16 = lane & 15;
    // B fragments: lane -> col n=l16, k = 16*half + e (contiguous)
    v16bf bfr[4];
    #pragma unroll
    for (int j = 0; j < 4; ++j) {
        const unsigned short* p = &Bl[(wn * 64 + j * 16 + l16) * 40 + half * 16];
        bfr[j] = ldfrag(p, p + 8);
    }
    // A fragments: lane -> row m=l16, k = 16*half + e (contiguous)
    #pragma unroll
    for (int i = 0; i < 4; ++i) {
        const unsigned short* pr = &Al[(wm * 64 + i * 16 + l16) * 40];
        v16bf af = ldfrag(pr + half * 16, pr + half * 16 + 8);
        #pragma unroll
        for (int j = 0; j < 4; ++j)
            acc[i][j] = wmma_bf16(af, bfr[j], acc[i][j]);
    }
}

// one pipeline step: finish stage(cur), prefetch into nxt, compute from cur
__device__ __forceinline__ void gemm_step(const unsigned short* __restrict__ A,
                                          const unsigned short* __restrict__ Wt,
                                          int m0, int n0, int kb_next, bool do_stage,
                                          unsigned short* Al_cur, unsigned short* Bl_cur,
                                          unsigned short* Al_nxt, unsigned short* Bl_nxt,
                                          v8f acc[4][4]) {
    async_join();
    __syncthreads();                 // stage(cur) complete; readers of nxt done
    if (do_stage)
        gemm_stage(A, Wt, m0, n0, kb_next, Al_nxt, Bl_nxt);
    gemm_compute(Al_cur, Bl_cur, acc);
}

// pipelined mainloop, unrolled x2 so buffer identity is compile-time (keeps
// LDS accesses in AS3 -> ds_load_b128, not flat_load via generic pointers)
__device__ __forceinline__ void gemm_main(const unsigned short* __restrict__ A,
                                          const unsigned short* __restrict__ Wt,
                                          int m0, int n0,
                                          unsigned short* Al0, unsigned short* Al1,
                                          unsigned short* Bl0, unsigned short* Bl1,
                                          v8f acc[4][4]) {
    gemm_stage(A, Wt, m0, n0, 0, Al0, Bl0);
    #pragma unroll 1
    for (int it = 0; it < 32; it += 2) {
        gemm_step(A, Wt, m0, n0, (it + 1) * 32, true,
                  Al0, Bl0, Al1, Bl1, acc);
        gemm_step(A, Wt, m0, n0, (it + 2) * 32, (it + 2) < 32,
                  Al1, Bl1, Al0, Bl0, acc);
    }
}

// ---------------- kernel 1: QKV projection ----------------
__global__ __launch_bounds__(256)
void qkv_gemm(const unsigned short* __restrict__ Xb,
              const unsigned short* __restrict__ Wt,   // [3072][1024]
              unsigned short* __restrict__ QKV)        // 3 x [B][H][T][D] bf16
{
    __shared__ __align__(16) unsigned short Al0[128 * 40], Al1[128 * 40];
    __shared__ __align__(16) unsigned short Bl0[256 * 40], Bl1[256 * 40];
    v8f acc[4][4];
    #pragma unroll
    for (int i = 0; i < 4; ++i)
        #pragma unroll
        for (int j = 0; j < 4; ++j) acc[i][j] = vzero();

    const int m0 = blockIdx.y * 128, n0 = blockIdx.x * 256;
    gemm_main(Xb, Wt, m0, n0, Al0, Al1, Bl0, Bl1, acc);

    const int lane = threadIdx.x & 31;
    const int wave = threadIdx.x >> 5;
    const int wm = wave >> 2, wn = wave & 3;
    const int half = lane >> 4, l16 = lane & 15;
    #pragma unroll
    for (int i = 0; i < 4; ++i) {
        #pragma unroll
        for (int j = 0; j < 4; ++j) {
            int n = n0 + wn * 64 + j * 16 + l16;
            int which = n >> 10;          // 0:q 1:k 2:v
            int c = n & 1023;
            int h = c >> 6, d = c & 63;
            #pragma unroll
            for (int v = 0; v < 8; ++v) {
                int m = m0 + wm * 64 + i * 16 + half * 8 + v;
                int b = m >> 11, t = m & 2047;
                size_t idx = (size_t)which * QKV1 +
                             ((size_t)((b * H_ + h) * T_ + t)) * D_ + d;
                QKV[idx] = f2bf(acc[i][j][v]);
            }
        }
    }
}

// ---------------- kernel 2: causal flash attention ----------------
// grid(T/64, B*H), 128 threads = 4 waves; wave w owns q rows [q0+16w, +16).
// S^T = K * Q^T: softmax lane-local; P A-fragment is elementwise from S^T accs.
__global__ __launch_bounds__(128)
void attn_kernel(const unsigned short* __restrict__ QKV,
                 unsigned short* __restrict__ AT)   // [B*T][C] bf16
{
    __shared__ __align__(16) unsigned short Klds[32 * 72];  // [k][d], stride 72
    __shared__ __align__(16) unsigned short Vt[64 * 40];    // [d][k], stride 40

    const int tid  = threadIdx.x;
    const int lane = tid & 31, wave = tid >> 5;
    const int half = lane >> 4, l16 = lane & 15;
    const int q0 = blockIdx.x * 64;
    const int qw = q0 + wave * 16;
    const int bh = blockIdx.y;

    const size_t base = (size_t)bh * T_ * D_;
    const unsigned short* Qp = QKV + base;
    const unsigned short* Kp = QKV + (size_t)QKV1 + base;
    const unsigned short* Vp = QKV + 2 * (size_t)QKV1 + base;

    // Q^T B-fragments in regs: lane -> col q=l16, d = 32c + 16*half + e
    v16bf qf[2];
    #pragma unroll
    for (int c = 0; c < 2; ++c) {
        const unsigned short* p = Qp + (size_t)(qw + l16) * 64 + 32 * c + 16 * half;
        qf[c] = ldfrag(p, p + 8);
    }

    v8f o[4];
    #pragma unroll
    for (int n = 0; n < 4; ++n) o[n] = vzero();
    float m_run = -1e30f, l_run = 0.f;

    const int krow = tid >> 2;           // 0..31
    const int kseg = (tid & 3) * 16;     // 0/16/32/48 ushorts
    const int kp   = tid >> 3;           // 0..15
    const int db   = (tid & 7) * 8;      // 0..56
    const float scale = 0.125f;          // 1/sqrt(64)

    #pragma unroll 1
    for (int kb = 0; kb < q0 + 64; kb += 32) {
        // stage K rows [kb, kb+32) row-major via async path
        {
            const unsigned short* g = Kp + (size_t)(kb + krow) * 64 + kseg;
            cp16(g,     &Klds[krow * 72 + kseg]);
            cp16(g + 8, &Klds[krow * 72 + kseg + 8]);
        }
        // stage V transposed: Vt[d][k], k-pairs packed as b32 (needs VALU)
        {
            const unsigned short* g0 = Vp + (size_t)(kb + 2 * kp) * 64 + db;
            const unsigned short* g1 = g0 + 64;
            union { u32x4 q; unsigned short u[8]; } r0, r1;
            r0.q = *(const u32x4*)g0;
            r1.q = *(const u32x4*)g1;
            #pragma unroll
            for (int i = 0; i < 8; ++i) {
                unsigned int pk = (unsigned int)r0.u[i] | ((unsigned int)r1.u[i] << 16);
                *(unsigned int*)&Vt[(db + i) * 40 + 2 * kp] = pk;
            }
        }
        async_join();
        __syncthreads();

        if (kb <= qw + 15) {           // wave-uniform: EXEC stays all-ones
            float p[2][8];
            float tmax = -1e30f;
            #pragma unroll
            for (int j = 0; j < 2; ++j) {
                v8f s = vzero();
                #pragma unroll
                for (int c = 0; c < 2; ++c) {
                    const unsigned short* pr = &Klds[(16 * j + l16) * 72 + 32 * c];
                    v16bf kf = ldfrag(pr + half * 16, pr + half * 16 + 8);
                    s = wmma_bf16(kf, qf[c], s);
                }
                #pragma unroll
                for (int v = 0; v < 8; ++v) {
                    int kg = kb + 16 * j + half * 8 + v;    // key index (S^T row)
                    int qg = qw + l16;                      // query index (S^T col)
                    float sv = (kg <= qg) ? s[v] * scale : -1e30f;
                    p[j][v] = sv;
                    tmax = fmaxf(tmax, sv);
                }
            }
            tmax = fmaxf(tmax, bcast(tmax, lane ^ 16));     // half-wave pair combine
            float m_new = fmaxf(m_run, tmax);
            float alpha = __expf(m_run - m_new);
            float rsum = 0.f;
            union { v16bf v; unsigned short u[16]; } pa;    // P A-fragment
            #pragma unroll
            for (int j = 0; j < 2; ++j)
                #pragma unroll
                for (int v = 0; v < 8; ++v) {
                    float e = __expf(p[j][v] - m_new);
                    rsum += e;
                    pa.u[j * 8 + v] = f2bf(e);              // exact A-layout mapping
                }
            rsum += bcast(rsum, lane ^ 16);
            l_run = l_run * alpha + rsum;
            m_run = m_new;
            // rescale O rows (row q = half*8+v; alpha lives in lane q)
            #pragma unroll
            for (int v = 0; v < 8; ++v) {
                float fac = bcast(alpha, half * 8 + v);
                #pragma unroll
                for (int n = 0; n < 4; ++n) o[n][v] *= fac;
            }
            // O += P * V
            #pragma unroll
            for (int n = 0; n < 4; ++n) {
                const unsigned short* pr = &Vt[(16 * n + l16) * 40];
                v16bf vf = ldfrag(pr + 8 * half, pr + 16 + 8 * half);
                o[n] = wmma_bf16(pa.v, vf, o[n]);
            }
        }
        __syncthreads();
    }

    const float inv = 1.0f / l_run;
    const int b = bh >> 4, h = bh & 15;
    #pragma unroll
    for (int v = 0; v < 8; ++v) {
        float fac = bcast(inv, half * 8 + v);
        int t = qw + half * 8 + v;
        size_t row = ((size_t)(b * T_) + t) * C_ + h * 64;
        #pragma unroll
        for (int n = 0; n < 4; ++n)
            AT[row + n * 16 + l16] = f2bf(o[n][v] * fac);
    }
}

// ---------------- kernel 3: output projection + bias ----------------
__global__ __launch_bounds__(256)
void out_gemm(const unsigned short* __restrict__ Ab,
              const unsigned short* __restrict__ Wot,   // [1024][1024] transposed
              const float* __restrict__ bo,
              float* __restrict__ Out)
{
    __shared__ __align__(16) unsigned short Al0[128 * 40], Al1[128 * 40];
    __shared__ __align__(16) unsigned short Bl0[256 * 40], Bl1[256 * 40];
    v8f acc[4][4];
    #pragma unroll
    for (int i = 0; i < 4; ++i)
        #pragma unroll
        for (int j = 0; j < 4; ++j) acc[i][j] = vzero();

    const int m0 = blockIdx.y * 128, n0 = blockIdx.x * 256;
    gemm_main(Ab, Wot, m0, n0, Al0, Al1, Bl0, Bl1, acc);

    const int lane = threadIdx.x & 31;
    const int wave = threadIdx.x >> 5;
    const int wm = wave >> 2, wn = wave & 3;
    const int half = lane >> 4, l16 = lane & 15;
    #pragma unroll
    for (int i = 0; i < 4; ++i) {
        #pragma unroll
        for (int j = 0; j < 4; ++j) {
            int n = n0 + wn * 64 + j * 16 + l16;
            float bias = bo[n];
            #pragma unroll
            for (int v = 0; v < 8; ++v) {
                int m = m0 + wm * 64 + i * 16 + half * 8 + v;
                Out[(size_t)m * C_ + n] = acc[i][j][v] + bias;
            }
        }
    }
}

// ---------------- host launcher ----------------
extern "C" void kernel_launch(void* const* d_in, const int* in_sizes, int n_in,
                              void* d_out, int out_size, void* d_ws, size_t ws_size,
                              hipStream_t stream) {
    const float* x    = (const float*)d_in[0];   // [B,T,C]
    const float* Wqkv = (const float*)d_in[1];   // [C,3C]
    const float* Wo   = (const float*)d_in[2];   // [C,C]
    const float* bo   = (const float*)d_in[3];   // [C]
    float* out = (float*)d_out;

    unsigned short* ws = (unsigned short*)d_ws;
    unsigned short* Xb    = ws;                               // [4096][1024]
    unsigned short* Wqkvt = ws + (size_t)MTOT * C_;           // [3072][1024] (transposed)
    unsigned short* Wot   = Wqkvt + (size_t)C_ * NQKV;        // [1024][1024] (transposed)
    unsigned short* QKV   = Wot + (size_t)C_ * C_;            // 3 x [B][H][T][D]
    unsigned short* AT    = QKV + 3 * (size_t)QKV1;           // [4096][1024]

    // bf16 conversion / weight transposes (bandwidth-trivial at 23.3 TB/s)
    cvt_f32_bf16 <<<1024, 256, 0, stream>>>(x, Xb, MTOT * C_);
    cvt_transpose<<<dim3(NQKV / 64, C_ / 64), 256, 0, stream>>>(Wqkv, Wqkvt, C_, NQKV);
    cvt_transpose<<<dim3(C_ / 64,   C_ / 64), 256, 0, stream>>>(Wo,   Wot,   C_, C_);

    // QKV projection: [4096x1024] x [1024x3072]
    qkv_gemm<<<dim3(NQKV / 256, MTOT / 128), 256, 0, stream>>>(Xb, Wqkvt, QKV);

    // causal flash attention
    attn_kernel<<<dim3(T_ / 64, B_ * H_), 128, 0, stream>>>(QKV, AT);

    // output projection: [4096x1024] x [1024x1024] + bias
    out_gemm<<<dim3(C_ / 256, MTOT / 128), 256, 0, stream>>>(AT, Wot, bo, out);
}
